// PointTransformerLayerv2_12309376270553
// MI455X (gfx1250) — matrix operand order repaired and checked by reading, not compiled, not run on hardware
//
#include <hip/hip_runtime.h>
#include <hip/hip_bf16.h>

#define DIM  128
#define NPTS 4096
#define NBAT 4
#define KNN  16

typedef __attribute__((ext_vector_type(16))) __bf16 v16bf;
typedef __attribute__((ext_vector_type(8)))  float  v8f;

struct alignas(16) U128 { unsigned int w[4]; };

// ---- A fragment (16x32 bf16, M rows across lanes %16) ----
// ISA layout: lanes 0-15: elems 0..7 -> K 0..7, elems 8..15 -> K 16..23
//             lanes16-31: elems 0..7 -> K 8..15, elems 8..15 -> K 24..31
__device__ __forceinline__ v16bf load_fragA(const __bf16* row, int s, int half) {
    const int k0 = 32 * s + (half ? 8  : 0);
    const int k1 = 32 * s + (half ? 24 : 16);
    v16bf v;
    *reinterpret_cast<U128*>(&v)       = *reinterpret_cast<const U128*>(row + k0);
    *(reinterpret_cast<U128*>(&v) + 1) = *reinterpret_cast<const U128*>(row + k1);
    return v;
}

// ---- B fragment (32x16 bf16, N cols across lanes %16) ----
// Same K layout as A: lanes 0-15: elems 0..7 -> K 0..7, elems 8..15 -> K 16..23
//                     lanes16-31: elems 0..7 -> K 8..15, elems 8..15 -> K 24..31
__device__ __forceinline__ v16bf load_fragB(const __bf16* row, int s, int half) {
    const int k0 = 32 * s + 8 * half;
    v16bf v;
    *reinterpret_cast<U128*>(&v)       = *reinterpret_cast<const U128*>(row + k0);
    *(reinterpret_cast<U128*>(&v) + 1) = *reinterpret_cast<const U128*>(row + k0 + 16);
    return v;
}

__device__ __forceinline__ v8f wmma_bf16(v16bf a, v16bf b, v8f c) {
    return __builtin_amdgcn_wmma_f32_16x16x32_bf16(false, a, false, b,
                                                   (short)0, c, false, false);
}

// ============================== kNN ==============================
__global__ void __launch_bounds__(256) knn_kernel(const float* __restrict__ pos,
                                                  int* __restrict__ idx) {
    __shared__ float sp[NPTS * 3];                       // 48 KB
    const int b   = blockIdx.y;
    const int tid = threadIdx.x;
    const float* pb = pos + (size_t)b * NPTS * 3;
    for (int i = tid; i < NPTS * 3; i += 256) sp[i] = pb[i];
    __syncthreads();

    const int n = blockIdx.x * 256 + tid;
    const float px = sp[n * 3 + 0], py = sp[n * 3 + 1], pz = sp[n * 3 + 2];

    float bd[KNN]; int bi[KNN];
#pragma unroll
    for (int j = 0; j < KNN; ++j) { bd[j] = 3.4e38f; bi[j] = 0; }

    for (int m = 0; m < NPTS; ++m) {
        const float dx = sp[m * 3 + 0] - px;
        const float dy = sp[m * 3 + 1] - py;
        const float dz = sp[m * 3 + 2] - pz;
        const float d  = dx * dx + dy * dy + dz * dz;    // ascending d == topk(pdist)
        if (d < bd[KNN - 1]) {
            bd[KNN - 1] = d; bi[KNN - 1] = m;
#pragma unroll
            for (int j = KNN - 1; j > 0; --j) {
                if (bd[j] < bd[j - 1]) {
                    float td = bd[j]; bd[j] = bd[j - 1]; bd[j - 1] = td;
                    int   ti = bi[j]; bi[j] = bi[j - 1]; bi[j - 1] = ti;
                }
            }
        }
    }
    int* op = idx + ((size_t)b * NPTS + n) * KNN;
#pragma unroll
    for (int j = 0; j < KNN; ++j) op[j] = bi[j];
}

// ========================= f32 -> bf16 ==========================
__global__ void cvt_bf16_kernel(const float* __restrict__ src,
                                __bf16* __restrict__ dst, int n) {
    const int i = blockIdx.x * 256 + threadIdx.x;
    if (i < n) dst[i] = (__bf16)src[i];
}

// ===================== fused transformer ========================
__global__ void __launch_bounds__(256) pt_fused_kernel(
    const float* __restrict__ x,       const float* __restrict__ pos,
    const float* __restrict__ pos_w1,  const float* __restrict__ pos_b1,
    const float* __restrict__ pos_b2,
    const float* __restrict__ attn_b1, const float* __restrict__ attn_b2,
    const float* __restrict__ wq,      const float* __restrict__ wo,
    const float* __restrict__ bo,
    const __bf16* __restrict__ xb,
    const __bf16* __restrict__ wkb,    const __bf16* __restrict__ wvb,
    const __bf16* __restrict__ pw2b,   const __bf16* __restrict__ aw1b,
    const __bf16* __restrict__ aw2b,
    const int* __restrict__ idx,       float* __restrict__ out)
{
    __shared__ __align__(16) __bf16 sb[8][16 * DIM];   // 32 KB: per-wave tile buffer
    __shared__ float sq[8][DIM];                       // 4 KB: per-wave q/out vector

    const int tid  = threadIdx.x;
    const int wave = tid >> 5, lane = tid & 31;
    const int half = lane >> 4, lm = lane & 15;
    const int gp   = blockIdx.x * 8 + wave;            // global point id
    const int b    = gp >> 12, n = gp & (NPTS - 1);

    __bf16* sbuf = sb[wave];
    float*  sqv  = sq[wave];

    const int* nbr  = idx + (size_t)gp * KNN;
    const int  myrow = nbr[lm];                        // neighbor for this lane's M row

    // --- A fragments of gathered neighbor features (16 x 128) ---
    const __bf16* featrow = xb + ((size_t)b * NPTS + myrow) * DIM;
    v16bf fA[4];
#pragma unroll
    for (int s = 0; s < 4; ++s) fA[s] = load_fragA(featrow, s, half);

    // --- q = x @ wq^T (1x128x128, VALU; 4 outputs per lane) ---
    {
        const float* xrow = x + ((size_t)b * NPTS + n) * DIM;
        float q0 = 0.f, q1 = 0.f, q2 = 0.f, q3 = 0.f;
        for (int i = 0; i < DIM; ++i) {
            const float xi = xrow[i];
            q0 += xi * wq[(lane      ) * DIM + i];
            q1 += xi * wq[(lane + 32 ) * DIM + i];
            q2 += xi * wq[(lane + 64 ) * DIM + i];
            q3 += xi * wq[(lane + 96 ) * DIM + i];
        }
        sqv[lane] = q0; sqv[lane + 32] = q1; sqv[lane + 64] = q2; sqv[lane + 96] = q3;
    }

    // --- pos-MLP layer 1: relu(rel @ pos_w1^T + b1) -> sbuf (16x128 bf16) ---
    {
        const float* pq = pos + ((size_t)b * NPTS + n     ) * 3;
        const float* pm = pos + ((size_t)b * NPTS + myrow ) * 3;
        const float rx = pm[0] - pq[0], ry = pm[1] - pq[1], rz = pm[2] - pq[2];
        const int o0 = 64 * half;
        for (int o = o0; o < o0 + 64; ++o) {
            float h = rx * pos_w1[o * 3 + 0] + ry * pos_w1[o * 3 + 1]
                    + rz * pos_w1[o * 3 + 2] + pos_b1[o];
            sbuf[lm * DIM + o] = (__bf16)fmaxf(h, 0.f);
        }
    }
    asm volatile("s_wait_dscnt 0" ::: "memory");       // wave-internal LDS fence

    v16bf h1A[4];
#pragma unroll
    for (int s = 0; s < 4; ++s) h1A[s] = load_fragA(sbuf + lm * DIM, s, half);
    asm volatile("" ::: "memory");

    // --- phase 2: k = feat@wk^T, pe = h1@pos_w2^T; energy = q - k + pe -> sbuf ---
    for (int t = 0; t < 8; ++t) {
        v8f kacc  = {0.f,0.f,0.f,0.f,0.f,0.f,0.f,0.f};
        v8f peacc = {0.f,0.f,0.f,0.f,0.f,0.f,0.f,0.f};
        const __bf16* wkrow = wkb  + (size_t)(16 * t + lm) * DIM;
        const __bf16* pwrow = pw2b + (size_t)(16 * t + lm) * DIM;
#pragma unroll
        for (int s = 0; s < 4; ++s) {
            kacc  = wmma_bf16(fA[s],  load_fragB(wkrow, s, half), kacc);
            peacc = wmma_bf16(h1A[s], load_fragB(pwrow, s, half), peacc);
        }
        const float qcol = sqv[16 * t + lm];
        const float pb2  = pos_b2[16 * t + lm];
#pragma unroll
        for (int r = 0; r < 8; ++r) {
            const float e = qcol - kacc[r] + peacc[r] + pb2;
            sbuf[(r + 8 * half) * DIM + 16 * t + lm] = (__bf16)e;
        }
    }
    asm volatile("s_wait_dscnt 0" ::: "memory");

    v16bf eA[4];
#pragma unroll
    for (int s = 0; s < 4; ++s) eA[s] = load_fragA(sbuf + lm * DIM, s, half);
    asm volatile("" ::: "memory");

    // --- phase 3: relu(energy @ attn_w1^T + b1) -> sbuf ---
    for (int t = 0; t < 8; ++t) {
        v8f h = {0.f,0.f,0.f,0.f,0.f,0.f,0.f,0.f};
        const __bf16* wrow = aw1b + (size_t)(16 * t + lm) * DIM;
#pragma unroll
        for (int s = 0; s < 4; ++s)
            h = wmma_bf16(eA[s], load_fragB(wrow, s, half), h);
        const float b1 = attn_b1[16 * t + lm];
#pragma unroll
        for (int r = 0; r < 8; ++r)
            sbuf[(r + 8 * half) * DIM + 16 * t + lm] = (__bf16)fmaxf(h[r] + b1, 0.f);
    }
    asm volatile("s_wait_dscnt 0" ::: "memory");

    v16bf hA[4];
#pragma unroll
    for (int s = 0; s < 4; ++s) hA[s] = load_fragA(sbuf + lm * DIM, s, half);
    asm volatile("" ::: "memory");

    // --- phase 4: attn = softmax_K(h @ attn_w2^T + b2); out = sum_K attn*v ---
    for (int t = 0; t < 8; ++t) {
        v8f aacc = {0.f,0.f,0.f,0.f,0.f,0.f,0.f,0.f};
        v8f vacc = {0.f,0.f,0.f,0.f,0.f,0.f,0.f,0.f};
        const __bf16* w2row = aw2b + (size_t)(16 * t + lm) * DIM;
        const __bf16* wvrow = wvb  + (size_t)(16 * t + lm) * DIM;
#pragma unroll
        for (int s = 0; s < 4; ++s) {
            aacc = wmma_bf16(hA[s], load_fragB(w2row, s, half), aacc);
            vacc = wmma_bf16(fA[s], load_fragB(wvrow, s, half), vacc);
        }
        const float b2 = attn_b2[16 * t + lm];
        float a[8], mx = -3.4e38f;
#pragma unroll
        for (int r = 0; r < 8; ++r) { a[r] = aacc[r] + b2; mx = fmaxf(mx, a[r]); }
        mx = fmaxf(mx, __shfl_xor(mx, 16, 32));        // combine lane halves (M 0-7 | 8-15)
        float sum = 0.f;
#pragma unroll
        for (int r = 0; r < 8; ++r) { a[r] = __expf(a[r] - mx); sum += a[r]; }
        sum += __shfl_xor(sum, 16, 32);
        const float inv = 1.f / sum;
        float part = 0.f;
#pragma unroll
        for (int r = 0; r < 8; ++r) part += a[r] * vacc[r];
        part *= inv;
        part += __shfl_xor(part, 16, 32);              // sum over all 16 neighbors
        sqv[16 * t + lm] = part;                       // both halves write same value
    }
    asm volatile("s_wait_dscnt 0" ::: "memory");

    // --- phase 5: y = out @ wo^T + bo ---
    {
        float y0 = bo[lane], y1 = bo[lane + 32], y2 = bo[lane + 64], y3 = bo[lane + 96];
        for (int i = 0; i < DIM; ++i) {
            const float oi = sqv[i];
            y0 += oi * wo[(lane      ) * DIM + i];
            y1 += oi * wo[(lane + 32 ) * DIM + i];
            y2 += oi * wo[(lane + 64 ) * DIM + i];
            y3 += oi * wo[(lane + 96 ) * DIM + i];
        }
        float* orow = out + ((size_t)b * NPTS + n) * DIM;
        orow[lane] = y0; orow[lane + 32] = y1; orow[lane + 64] = y2; orow[lane + 96] = y3;
    }
}

// ============================ launch ============================
extern "C" void kernel_launch(void* const* d_in, const int* in_sizes, int n_in,
                              void* d_out, int out_size, void* d_ws, size_t ws_size,
                              hipStream_t stream) {
    const float* x       = (const float*)d_in[0];
    const float* pos     = (const float*)d_in[1];
    const float* pos_w1  = (const float*)d_in[2];
    const float* pos_b1  = (const float*)d_in[3];
    const float* pos_w2  = (const float*)d_in[4];
    const float* pos_b2  = (const float*)d_in[5];
    const float* attn_w1 = (const float*)d_in[6];
    const float* attn_b1 = (const float*)d_in[7];
    const float* attn_w2 = (const float*)d_in[8];
    const float* attn_b2 = (const float*)d_in[9];
    const float* wq      = (const float*)d_in[10];
    const float* wk      = (const float*)d_in[11];
    const float* wv      = (const float*)d_in[12];
    const float* wo      = (const float*)d_in[13];
    const float* bo      = (const float*)d_in[14];
    float* out = (float*)d_out;

    char* ws = (char*)d_ws;
    const size_t NX = (size_t)NBAT * NPTS * DIM;       // 2,097,152
    const size_t NW = (size_t)DIM * DIM;               // 16,384
    int*    idx  = (int*)ws;                                        // 1 MB
    __bf16* xb   = (__bf16*)(ws + 1048576);                         // 4 MB
    __bf16* wkb  = (__bf16*)(ws + 1048576 + NX * 2);
    __bf16* wvb  = wkb  + NW;
    __bf16* pw2b = wvb  + NW;
    __bf16* aw1b = pw2b + NW;
    __bf16* aw2b = aw1b + NW;

    knn_kernel<<<dim3(NPTS / 256, NBAT), 256, 0, stream>>>(pos, idx);

    cvt_bf16_kernel<<<(int)((NX + 255) / 256), 256, 0, stream>>>(x, xb, (int)NX);
    cvt_bf16_kernel<<<(int)((NW + 255) / 256), 256, 0, stream>>>(wk,      wkb,  (int)NW);
    cvt_bf16_kernel<<<(int)((NW + 255) / 256), 256, 0, stream>>>(wv,      wvb,  (int)NW);
    cvt_bf16_kernel<<<(int)((NW + 255) / 256), 256, 0, stream>>>(pos_w2,  pw2b, (int)NW);
    cvt_bf16_kernel<<<(int)((NW + 255) / 256), 256, 0, stream>>>(attn_w1, aw1b, (int)NW);
    cvt_bf16_kernel<<<(int)((NW + 255) / 256), 256, 0, stream>>>(attn_w2, aw2b, (int)NW);

    pt_fused_kernel<<<(NBAT * NPTS) / 8, 256, 0, stream>>>(
        x, pos, pos_w1, pos_b1, pos_b2, attn_b1, attn_b2, wq, wo, bo,
        xb, wkb, wvb, pw2b, aw1b, aw2b, idx, out);
}
